// MultiHeadAttention_2319282339926
// MI455X (gfx1250) — matrix operand order repaired and checked by reading, not compiled, not run on hardware
//
#include <hip/hip_runtime.h>

// ---------------------------------------------------------------------------
// Types for gfx1250 WMMA (wave32)
// ---------------------------------------------------------------------------
typedef __bf16 bf16;
typedef __attribute__((ext_vector_type(16))) __bf16 bf16x16;
typedef __attribute__((ext_vector_type(8)))  __bf16 bf16x8;
typedef __attribute__((ext_vector_type(8)))  float  f32x8;
typedef __attribute__((ext_vector_type(4)))  float  f32x4;

#define BATCH 2
#define SEQ   2048
#define DMODEL 1024
#define NHEAD 16
#define HDIM  64
#define GM (BATCH * SEQ)   // 4096 rows
#define GN DMODEL          // 1024 cols
#define GK DMODEL          // 1024 reduction

__device__ __forceinline__ f32x8 wmma_bf16(bf16x16 a, bf16x16 b, f32x8 c) {
  return __builtin_amdgcn_wmma_f32_16x16x32_bf16(
      /*neg_a=*/false, a, /*neg_b=*/false, b,
      /*c_mod=*/(short)0, c, /*reuse_a=*/false, /*reuse_b=*/false);
}

// A-fragment (16x32 bf16) built from an fp32 row.
// elements 0..15 -> K = kbase + 2*cc + i      (cc = lane<16 ? 0 : 8)
__device__ __forceinline__ bf16x16 make_afrag_f32(const float* rowp, int kbase, int cc) {
  const float* p0 = rowp + kbase + 2 * cc;
  f32x4 a = *(const f32x4*)(p0);
  f32x4 b = *(const f32x4*)(p0 + 4);
  f32x4 c = *(const f32x4*)(p0 + 8);
  f32x4 d = *(const f32x4*)(p0 + 12);
  bf16x16 r;
#pragma unroll
  for (int i = 0; i < 4; ++i) {
    r[i]      = (bf16)a[i];
    r[i + 4]  = (bf16)b[i];
    r[i + 8]  = (bf16)c[i];
    r[i + 12] = (bf16)d[i];
  }
  return r;
}

// A-fragment from a bf16 row (one contiguous 32B run at K = 2*cc)
__device__ __forceinline__ bf16x16 make_afrag_bf16(const bf16* rowp, int cc) {
  bf16x8 lo = *(const bf16x8*)(rowp + 2 * cc);
  bf16x8 hi = *(const bf16x8*)(rowp + 2 * cc + 8);
  bf16x16 r;
#pragma unroll
  for (int i = 0; i < 8; ++i) { r[i] = lo[i]; r[i + 8] = hi[i]; }
  return r;
}

// B-fragment (32x16 bf16): lane = column, K = (lane<16?0:16)+i, contiguous 32B
__device__ __forceinline__ bf16x16 load_bfrag(const bf16* p) {
  return *(const bf16x16*)p;
}

// ---------------------------------------------------------------------------
// fp32 -> bf16 convert (weights)
// ---------------------------------------------------------------------------
__global__ void cvt_f32_bf16_kernel(const float* __restrict__ x,
                                    bf16* __restrict__ y, int n) {
  int i = blockIdx.x * blockDim.x + threadIdx.x;
  if (i < n) y[i] = (bf16)x[i];
}

// ---------------------------------------------------------------------------
// GEMM: out = A(fp32)[GMxGK] * W(bf16)[GNxGK]^T + bias
// MODE 0: bf16 row-major            (K projection)
// MODE 1: bf16 row-major * scale    (Q projection, pre-scaled by 1/sqrt(HD))
// MODE 2: bf16 transposed per head  (V -> Vt[b][h][d][s])
// MODE 3: fp32 row-major            (final output projection)
// Block: 256 threads = 8 waves (4 M-tiles x 2 N-tiles); wave tile 32x32.
// ---------------------------------------------------------------------------
template <int MODE>
__global__ __launch_bounds__(256) void gemm_bias_kernel(
    const float* __restrict__ A, const bf16* __restrict__ W,
    const float* __restrict__ bias, void* __restrict__ outv, float scale) {
  const int lane = threadIdx.x & 31;
  const int w    = threadIdx.x >> 5;
  const int m0   = blockIdx.x * 128 + (w & 3) * 32;
  const int n0   = blockIdx.y * 64  + (w >> 2) * 32;
  const int l16  = lane & 15;
  const bool hiL = lane >= 16;
  const int cc   = hiL ? 8 : 0;
  const int kb16 = hiL ? 16 : 0;

  const float* arow0 = A + (size_t)(m0 + l16) * GK;
  const float* arow1 = A + (size_t)(m0 + 16 + l16) * GK;
  const bf16*  wrow0 = W + (size_t)(n0 + l16) * GK + kb16;
  const bf16*  wrow1 = W + (size_t)(n0 + 16 + l16) * GK + kb16;

  f32x8 acc00 = {}, acc01 = {}, acc10 = {}, acc11 = {};
#pragma unroll 4
  for (int k0 = 0; k0 < GK; k0 += 32) {
    bf16x16 a0 = make_afrag_f32(arow0, k0, cc);
    bf16x16 a1 = make_afrag_f32(arow1, k0, cc);
    bf16x16 b0 = load_bfrag(wrow0 + k0);
    bf16x16 b1 = load_bfrag(wrow1 + k0);
    acc00 = wmma_bf16(a0, b0, acc00);
    acc01 = wmma_bf16(a0, b1, acc01);
    acc10 = wmma_bf16(a1, b0, acc10);
    acc11 = wmma_bf16(a1, b1, acc11);
  }

#pragma unroll
  for (int tm = 0; tm < 2; ++tm) {
#pragma unroll
    for (int tn = 0; tn < 2; ++tn) {
      const f32x8& acc = (tm == 0) ? (tn == 0 ? acc00 : acc01)
                                   : (tn == 0 ? acc10 : acc11);
#pragma unroll
      for (int r = 0; r < 8; ++r) {
        int m = m0 + tm * 16 + r + (hiL ? 8 : 0);
        int n = n0 + tn * 16 + l16;
        float v = acc[r] + bias[n];
        if (MODE == 1) v *= scale;
        if (MODE == 0 || MODE == 1) {
          ((bf16*)outv)[(size_t)m * GN + n] = (bf16)v;
        } else if (MODE == 3) {
          ((float*)outv)[(size_t)m * GN + n] = v;
        } else {  // MODE 2: Vt[((b*H + h)*HD + d)*S + s]
          int b = m >> 11, s = m & (SEQ - 1);
          int h = n >> 6, d = n & (HDIM - 1);
          ((bf16*)outv)[(((size_t)b * NHEAD + h) * HDIM + d) * SEQ + s] = (bf16)v;
        }
      }
    }
  }
}

// ---------------------------------------------------------------------------
// One 64-key flash-attention block.  MASKED=false for blocks fully below the
// causal diagonal (no compares/selects at all); MASKED=true for the single
// diagonal block.
// ---------------------------------------------------------------------------
template <bool MASKED>
__device__ __forceinline__ void attn_block(
    int kt, int q0, const bf16* __restrict__ Kbase, const bf16* __restrict__ Vbase,
    bf16x16 aq0, bf16x16 aq1, f32x8 (&acc)[4], float (&mrow)[8], float (&lrow)[8],
    bf16* __restrict__ ldsPw, int l16, bool hiL, int cc, int kb16) {
  // ---- scores: four 16x16 tiles over 64 key positions --------------------
  f32x8 s[4];
#pragma unroll
  for (int j = 0; j < 4; ++j) {
    const bf16* krow = Kbase + (size_t)(kt + j * 16 + l16) * DMODEL;
    f32x8 sj = {};
    sj = wmma_bf16(aq0, load_bfrag(krow + kb16), sj);
    sj = wmma_bf16(aq1, load_bfrag(krow + 32 + kb16), sj);
    s[j] = sj;
  }

  // ---- online softmax update (row stats live per 16-lane group) ----------
#pragma unroll
  for (int r = 0; r < 8; ++r) {
    float v[4];
    if (MASKED) {
      int qg = q0 + r + (hiL ? 8 : 0);
#pragma unroll
      for (int j = 0; j < 4; ++j) {
        int kg = kt + j * 16 + l16;
        v[j] = (kg <= qg) ? s[j][r] : -1e30f;
      }
    } else {
#pragma unroll
      for (int j = 0; j < 4; ++j) v[j] = s[j][r];
    }
    float tmax = fmaxf(fmaxf(v[0], v[1]), fmaxf(v[2], v[3]));
#pragma unroll
    for (int off = 8; off >= 1; off >>= 1)
      tmax = fmaxf(tmax, __shfl_xor(tmax, off, 32));
    float mnew = fmaxf(mrow[r], tmax);
    float sc   = __expf(mrow[r] - mnew);
    float p[4];
#pragma unroll
    for (int j = 0; j < 4; ++j) p[j] = __expf(v[j] - mnew);
    float rsum = (p[0] + p[1]) + (p[2] + p[3]);
#pragma unroll
    for (int off = 8; off >= 1; off >>= 1)
      rsum += __shfl_xor(rsum, off, 32);
    lrow[r] = lrow[r] * sc + rsum;
    mrow[r] = mnew;
#pragma unroll
    for (int t = 0; t < 4; ++t) acc[t][r] *= sc;
    // C-layout -> LDS (row-major 16x64 bf16 P tile)
    int prow = r + (hiL ? 8 : 0);
#pragma unroll
    for (int j = 0; j < 4; ++j)
      ldsPw[prow * 64 + j * 16 + l16] = (bf16)p[j];
  }
  // Wave-private LDS round-trip: wait for our DS stores (no block barrier —
  // waves in the block have different trip counts).
  asm volatile("s_wait_dscnt 0" ::: "memory");

  // ---- P (16x64) as two A-fragments from LDS, then P·V -------------------
  bf16x16 ap0 = make_afrag_bf16(&ldsPw[l16 * 64], cc);       // k 0..31
  bf16x16 ap1 = make_afrag_bf16(&ldsPw[l16 * 64 + 32], cc);  // k 32..63
#pragma unroll
  for (int t = 0; t < 4; ++t) {
    const bf16* vrow = Vbase + (size_t)(t * 16 + l16) * SEQ + kt + kb16;
    acc[t] = wmma_bf16(ap0, load_bfrag(vrow), acc[t]);
    acc[t] = wmma_bf16(ap1, load_bfrag(vrow + 32), acc[t]);
  }
}

// ---------------------------------------------------------------------------
// Causal flash attention.  One wave per (b, h, 16-query-row tile).
// Q pre-scaled by 1/sqrt(HD).  K stored [B,S,D] bf16, V stored [B,H,HD,S] bf16.
// Output ctx fp32 [B,S,D].
// ---------------------------------------------------------------------------
__global__ __launch_bounds__(256) void flash_attn_kernel(
    const bf16* __restrict__ Q, const bf16* __restrict__ Kb,
    const bf16* __restrict__ Vt, float* __restrict__ Ctx) {
  __shared__ bf16 ldsP[8][16 * 64];  // per-wave 16x64 P tile (16 KB total)

  const int lane = threadIdx.x & 31;
  const int w    = threadIdx.x >> 5;
  const int wid  = blockIdx.x * 8 + w;           // 0 .. 4095
  const int qt   = wid & 127;                    // S/16 tiles
  const int h    = (wid >> 7) & (NHEAD - 1);
  const int b    = wid >> 11;
  const int q0   = qt * 16;
  const int l16  = lane & 15;
  const bool hiL = lane >= 16;
  const int cc   = hiL ? 8 : 0;
  const int kb16 = hiL ? 16 : 0;
  bf16* ldsPw    = &ldsP[w][0];

  // Q A-fragments for the two d-steps (d 0..31 and 32..63)
  const bf16* qrow = Q + ((size_t)b * SEQ + q0 + l16) * DMODEL + h * HDIM;
  bf16x16 aq0 = make_afrag_bf16(qrow, cc);        // d offsets 0..31
  bf16x16 aq1 = make_afrag_bf16(qrow + 32, cc);   // d offsets 32..63

  const bf16* Kbase = Kb + (size_t)b * SEQ * DMODEL + h * HDIM;
  const bf16* Vbase = Vt + ((size_t)b * NHEAD + h) * HDIM * SEQ;

  f32x8 acc[4] = {};
  float mrow[8], lrow[8];
#pragma unroll
  for (int r = 0; r < 8; ++r) { mrow[r] = -1e30f; lrow[r] = 0.0f; }

  // Blocks fully below the diagonal need no causal masking:
  // block [kt, kt+63] is unmasked iff kt+63 <= q0  =>  nfull = (q0+1)>>6.
  const int nfull = (q0 + 1) >> 6;
  int kt = 0;
  for (int i = 0; i < nfull; ++i, kt += 64)
    attn_block<false>(kt, q0, Kbase, Vbase, aq0, aq1, acc, mrow, lrow,
                      ldsPw, l16, hiL, cc, kb16);
  // Exactly one diagonal (masked) block remains: kt <= q0+15 < kt+64.
  attn_block<true>(kt, q0, Kbase, Vbase, aq0, aq1, acc, mrow, lrow,
                   ldsPw, l16, hiL, cc, kb16);

  // ---- normalize and write fp32 context [B,S,D] --------------------------
#pragma unroll
  for (int r = 0; r < 8; ++r) {
    float inv = 1.0f / lrow[r];
    int m = q0 + r + (hiL ? 8 : 0);
    float* crow = Ctx + ((size_t)b * SEQ + m) * DMODEL + h * HDIM;
#pragma unroll
    for (int t = 0; t < 4; ++t) crow[t * 16 + l16] = acc[t][r] * inv;
  }
}

// ---------------------------------------------------------------------------
// Host launcher
// ---------------------------------------------------------------------------
extern "C" void kernel_launch(void* const* d_in, const int* in_sizes, int n_in,
                              void* d_out, int out_size, void* d_ws, size_t ws_size,
                              hipStream_t stream) {
  const float* query = (const float*)d_in[0];
  const float* key_  = (const float*)d_in[1];
  const float* value = (const float*)d_in[2];
  // d_in[3] = mask : causal, known statically — unused
  const float* Wq = (const float*)d_in[4];
  const float* bq = (const float*)d_in[5];
  const float* Wk = (const float*)d_in[6];
  const float* bk = (const float*)d_in[7];
  const float* Wv = (const float*)d_in[8];
  const float* bv = (const float*)d_in[9];
  const float* Wo = (const float*)d_in[10];
  const float* bo = (const float*)d_in[11];

  char* ws = (char*)d_ws;
  const size_t WBYTES = (size_t)DMODEL * DMODEL * sizeof(bf16);      // 2 MB
  const size_t XBYTES = (size_t)GM * DMODEL * sizeof(bf16);          // 8 MB
  bf16* Wq_b = (bf16*)(ws + 0 * WBYTES);
  bf16* Wk_b = (bf16*)(ws + 1 * WBYTES);
  bf16* Wv_b = (bf16*)(ws + 2 * WBYTES);
  bf16* Wo_b = (bf16*)(ws + 3 * WBYTES);
  bf16* Qbuf = (bf16*)(ws + 4 * WBYTES);
  bf16* Kbuf = (bf16*)(ws + 4 * WBYTES + 1 * XBYTES);
  bf16* Vtbf = (bf16*)(ws + 4 * WBYTES + 2 * XBYTES);
  float* Ctx = (float*)(ws + 4 * WBYTES + 3 * XBYTES);               // 16 MB

  const int NW = DMODEL * DMODEL;  // 1M elements per weight
  cvt_f32_bf16_kernel<<<NW / 256, 256, 0, stream>>>(Wq, Wq_b, NW);
  cvt_f32_bf16_kernel<<<NW / 256, 256, 0, stream>>>(Wk, Wk_b, NW);
  cvt_f32_bf16_kernel<<<NW / 256, 256, 0, stream>>>(Wv, Wv_b, NW);
  cvt_f32_bf16_kernel<<<NW / 256, 256, 0, stream>>>(Wo, Wo_b, NW);

  dim3 ggemm(GM / 128, GN / 64);  // (32, 16)
  const float qscale = 0.125f;    // 1/sqrt(HD=64)
  gemm_bias_kernel<1><<<ggemm, 256, 0, stream>>>(query, Wq_b, bq, Qbuf, qscale);
  gemm_bias_kernel<0><<<ggemm, 256, 0, stream>>>(key_,  Wk_b, bk, Kbuf, 1.0f);
  gemm_bias_kernel<2><<<ggemm, 256, 0, stream>>>(value, Wv_b, bv, Vtbf, 1.0f);

  // B*H*(S/16) = 4096 waves / 8 waves per block = 512 blocks
  flash_attn_kernel<<<512, 256, 0, stream>>>(Qbuf, Kbuf, Vtbf, Ctx);

  gemm_bias_kernel<3><<<ggemm, 256, 0, stream>>>(Ctx, Wo_b, bo, d_out, 1.0f);
}
